// EARLIEST_35768487641803
// MI455X (gfx1250) — matrix operand order repaired and checked by reading, not compiled, not run on hardware
//
#include <hip/hip_runtime.h>
#include <hip/hip_bf16.h>

// Problem constants (from reference): T=128, B=512, D=768, H=1024, C=1
#define T_STEPS 128
#define BATCH   512
#define D_IN    768
#define H_DIM   1024
#define G_DIM   4096   // 4*H

typedef unsigned short u16;
typedef __attribute__((ext_vector_type(16))) __bf16 v16bf;
typedef __attribute__((ext_vector_type(8)))  float  v8f;

union FragBF { uint4 q[2]; v16bf v; };

__device__ __forceinline__ u16 f32_to_bf16_rne(float f) {
    unsigned int u = __float_as_uint(f);
    u = u + 0x7FFFu + ((u >> 16) & 1u);
    return (u16)(u >> 16);
}

__device__ __forceinline__ float sigmoidf_(float x) {
    return 1.0f / (1.0f + __expf(-x));
}

// ---------------------------------------------------------------------------
// Generic fp32 -> bf16 conversion
// ---------------------------------------------------------------------------
__global__ __launch_bounds__(256) void cvt_f32_bf16_kernel(
    const float* __restrict__ in, u16* __restrict__ out, long n) {
    long i = (long)blockIdx.x * blockDim.x + threadIdx.x;
    if (i < n) out[i] = f32_to_bf16_rne(in[i]);
}

// bias = b_ih + b_hh (4H)
__global__ __launch_bounds__(256) void bias_sum_kernel(
    const float* __restrict__ bih, const float* __restrict__ bhh,
    float* __restrict__ bias) {
    int i = blockIdx.x * blockDim.x + threadIdx.x;
    if (i < G_DIM) bias[i] = bih[i] + bhh[i];
}

// Zero h/c/h_bf16, preds=0, hp=-1
__global__ __launch_bounds__(256) void init_state_kernel(
    float* __restrict__ h, float* __restrict__ c, u16* __restrict__ hb,
    float* __restrict__ preds, float* __restrict__ hp) {
    int i = blockIdx.x * blockDim.x + threadIdx.x;   // < BATCH*H_DIM
    h[i] = 0.0f; c[i] = 0.0f; hb[i] = 0;
    if (i < BATCH) { preds[i] = 0.0f; hp[i] = -1.0f; }
}

// ---------------------------------------------------------------------------
// Gates GEMM: gates(B x 4H) = x_t(B x D) @ W_ih^T + h(B x H) @ W_hh^T + bias
// bf16 WMMA 16x16x32, fp32 accumulate.
// One wave computes a 32x64 output tile: 2 A-frags x 4 B-frags -> 8 accums.
// This halves L2 weight-fragment traffic vs a 16x64 tile and gives the
// scheduler 8 independent WMMA chains to issue under outstanding loads.
// A/B fragments loaded directly from global memory in a contiguous-K wave32
// VGPR layout (not the ISA table order; valid as A and B match):
//   A 16x32 bf16: lane l<16 holds row M=l, K[0..15]; l>=16 holds
//                 K[16..31] (contiguous, matching B) -> two b128 loads per lane.
//   B 32x16 bf16: lane holds column N=l%16; l<16 -> K[0..15], l>=16 ->
//                 K[16..31], contiguous in K (W is row-major over K)
//                 -> two b128 loads per lane.
// Grid: 1024 waves = 16 M-tiles x 64 N-groups; 8 waves/block -> 128 blocks.
// ---------------------------------------------------------------------------
__global__ __launch_bounds__(256) void gates_gemm_kernel(
    const u16* __restrict__ A0,   // x_t  bf16 (B x D)
    const u16* __restrict__ A1,   // h    bf16 (B x H)
    const u16* __restrict__ W0,   // W_ih bf16 (4H x D)
    const u16* __restrict__ W1,   // W_hh bf16 (4H x H)
    const float* __restrict__ bias,
    float* __restrict__ gates)    // (B x 4H)
{
    const int lane  = threadIdx.x & 31;
    const int l16   = lane & 15;
    const int hi    = lane >> 4;                       // 0 or 1
    const int wid   = blockIdx.x * 8 + (threadIdx.x >> 5);
    const int tileM = (wid >> 6) << 5;                 // 0..480 (32-row tiles)
    const int baseN = (wid & 63) << 6;                 // 0..4032 (64-col groups)

    v8f acc0[4] = {v8f{}, v8f{}, v8f{}, v8f{}};        // rows tileM..tileM+15
    v8f acc1[4] = {v8f{}, v8f{}, v8f{}, v8f{}};        // rows tileM+16..tileM+31

    #pragma unroll
    for (int pass = 0; pass < 2; ++pass) {
        const u16* A = pass ? A1 : A0;
        const u16* W = pass ? W1 : W0;
        const int  K = pass ? H_DIM : D_IN;

        const u16* arow0 = A + (size_t)(tileM + l16) * K;
        const u16* arow1 = A + (size_t)(tileM + 16 + l16) * K;
        const u16* wrow[4];
        #pragma unroll
        for (int nb = 0; nb < 4; ++nb)
            wrow[nb] = W + (size_t)(baseN + nb * 16 + l16) * K;

        for (int k0 = 0; k0 < K; k0 += 32) {
            FragBF a0, a1, b[4];
            // A fragments (16x32 bf16 each)
            a0.q[0] = *(const uint4*)(arow0 + k0 + hi * 16);
            a0.q[1] = *(const uint4*)(arow0 + k0 + hi * 16 + 8);
            a1.q[0] = *(const uint4*)(arow1 + k0 + hi * 16);
            a1.q[1] = *(const uint4*)(arow1 + k0 + hi * 16 + 8);
            // B fragments (32x16 bf16): 16 contiguous K values per lane
            const int kb = k0 + hi * 16;
            #pragma unroll
            for (int nb = 0; nb < 4; ++nb) {
                b[nb].q[0] = *(const uint4*)(wrow[nb] + kb);
                b[nb].q[1] = *(const uint4*)(wrow[nb] + kb + 8);
            }
            #pragma unroll
            for (int nb = 0; nb < 4; ++nb) {
                acc0[nb] = __builtin_amdgcn_wmma_f32_16x16x32_bf16(
                    false, a0.v, false, b[nb].v, (short)0, acc0[nb], false, false);
                acc1[nb] = __builtin_amdgcn_wmma_f32_16x16x32_bf16(
                    false, a1.v, false, b[nb].v, (short)0, acc1[nb], false, false);
            }
        }
    }

    // C/D layout: VGPR r, lane l -> row M = r + (l>=16)*8, col N = l%16
    float bia[4];
    #pragma unroll
    for (int nb = 0; nb < 4; ++nb) bia[nb] = bias[baseN + nb * 16 + l16];

    #pragma unroll
    for (int r = 0; r < 8; ++r) {
        const int row0 = tileM + r + hi * 8;
        const int row1 = row0 + 16;
        float* g0 = gates + (size_t)row0 * G_DIM + baseN + l16;
        float* g1 = gates + (size_t)row1 * G_DIM + baseN + l16;
        #pragma unroll
        for (int nb = 0; nb < 4; ++nb) {
            g0[nb * 16] = acc0[nb][r] + bia[nb];
            g1[nb * 16] = acc1[nb][r] + bia[nb];
        }
    }
}

// ---------------------------------------------------------------------------
// LSTM cell elementwise (torch gate order i,f,g,o) + h -> bf16 repack
// ---------------------------------------------------------------------------
__global__ __launch_bounds__(256) void lstm_cell_kernel(
    const float* __restrict__ gates, float* __restrict__ c,
    float* __restrict__ h, u16* __restrict__ hb)
{
    int idx = blockIdx.x * blockDim.x + threadIdx.x;  // < BATCH*H_DIM
    int b = idx >> 10;
    int j = idx & (H_DIM - 1);
    const float* g = gates + (size_t)b * G_DIM;
    float gi = g[j];
    float gf = g[H_DIM + j];
    float gg = g[2 * H_DIM + j];
    float go = g[3 * H_DIM + j];
    float cn = sigmoidf_(gf) * c[idx] + sigmoidf_(gi) * tanhf(gg);
    float hn = sigmoidf_(go) * tanhf(cn);
    c[idx] = cn;
    h[idx] = hn;
    hb[idx] = f32_to_bf16_rne(hn);
}

// ---------------------------------------------------------------------------
// Controller: one wave per batch element. Two H-length dots (logits + ctrl),
// wave32 shuffle reduction, then halting-state update by lane 0.
// ---------------------------------------------------------------------------
__global__ __launch_bounds__(256) void controller_kernel(
    const float* __restrict__ h, const float* __restrict__ W_out,
    const float* __restrict__ b_out, const float* __restrict__ W_ctrl,
    const float* __restrict__ b_ctrl, const float* __restrict__ u,
    const float* __restrict__ eps, float* __restrict__ preds,
    float* __restrict__ hp, float* __restrict__ flog, int t)
{
    const int lane = threadIdx.x & 31;
    const int b = blockIdx.x * 8 + (threadIdx.x >> 5);
    const float* hb = h + (size_t)b * H_DIM;

    float d1 = 0.0f, d2 = 0.0f;
    #pragma unroll 4
    for (int k = lane; k < H_DIM; k += 32) {
        float hv = hb[k];
        d1 += hv * W_out[k];
        d2 += hv * W_ctrl[k];
    }
    #pragma unroll
    for (int off = 16; off > 0; off >>= 1) {
        d1 += __shfl_xor(d1, off, 32);
        d2 += __shfl_xor(d2, off, 32);
    }

    if (lane == 0) {
        const float tf = (float)t;
        const float logit = d1 + b_out[0];
        const float ctrl  = d2 + W_ctrl[H_DIM] * tf + b_ctrl[0];
        const float e = eps[0];
        float p = sigmoidf_(ctrl);
        p = (1.0f - e) * p + e * 0.05f;
        if (fabsf(p) <= 1e-8f) p += 1e-6f;               // jnp.isclose(p, 0)
        const float a = (u[(size_t)t * BATCH + b] < p) ? 1.0f : 0.0f;
        if (a == 1.0f && preds[b] == 0.0f) preds[b] = logit;
        if (hp[b] == -1.0f && a == 1.0f)   hp[b]    = tf;
        if (t == T_STEPS - 1)              flog[b]  = logit;
    }
}

// ---------------------------------------------------------------------------
// Finalize: resolve preds/halting points, compute mean.
// Output layout: logits[512] | halting_points[512] | halting_points_mean[1]
// ---------------------------------------------------------------------------
__global__ __launch_bounds__(512) void finalize_kernel(
    const float* __restrict__ preds, const float* __restrict__ hp,
    const float* __restrict__ flog, float* __restrict__ out)
{
    __shared__ float s[BATCH];
    const int b = threadIdx.x;
    const float hpr = (hp[b] == -1.0f) ? (float)(T_STEPS - 1) : hp[b];
    out[b] = (preds[b] == 0.0f) ? flog[b] : preds[b];
    out[BATCH + b] = hpr + 1.0f;
    s[b] = hpr + 1.0f;
    __syncthreads();
    #pragma unroll
    for (int off = BATCH / 2; off > 0; off >>= 1) {
        if (b < off) s[b] += s[b + off];
        __syncthreads();
    }
    if (b == 0) out[2 * BATCH] = (s[0] / (float)BATCH) / (float)(T_STEPS + 1);
}

// ---------------------------------------------------------------------------
// Host-side launch
// ---------------------------------------------------------------------------
static inline size_t align256(size_t x) { return (x + 255) & ~(size_t)255; }

extern "C" void kernel_launch(void* const* d_in, const int* in_sizes, int n_in,
                              void* d_out, int out_size, void* d_ws, size_t ws_size,
                              hipStream_t stream) {
    (void)in_sizes; (void)n_in; (void)out_size;
    const float* x      = (const float*)d_in[0];   // (T,B,D)
    const float* u      = (const float*)d_in[1];   // (T,B)
    const float* eps    = (const float*)d_in[2];   // (1,)
    const float* W_ih   = (const float*)d_in[3];   // (4H,D)
    const float* W_hh   = (const float*)d_in[4];   // (4H,H)
    const float* b_ih   = (const float*)d_in[5];   // (4H,)
    const float* b_hh   = (const float*)d_in[6];   // (4H,)
    const float* W_out  = (const float*)d_in[7];   // (1,H)
    const float* b_out  = (const float*)d_in[8];   // (1,)
    const float* W_ctrl = (const float*)d_in[9];   // (1,H+1)
    const float* b_ctrl = (const float*)d_in[10];  // (1,)
    // d_in[11] W_base, d_in[12] b_base: not needed for the returned outputs.

    char* ws = (char*)d_ws;
    size_t off = 0;
    u16*  Wih_bf = (u16*)(ws + off); off = align256(off + (size_t)G_DIM * D_IN * 2);
    u16*  Whh_bf = (u16*)(ws + off); off = align256(off + (size_t)G_DIM * H_DIM * 2);
    u16*  hb     = (u16*)(ws + off); off = align256(off + (size_t)BATCH * H_DIM * 2);
    float* bias  = (float*)(ws + off); off = align256(off + (size_t)G_DIM * 4);
    float* hbuf  = (float*)(ws + off); off = align256(off + (size_t)BATCH * H_DIM * 4);
    float* cbuf  = (float*)(ws + off); off = align256(off + (size_t)BATCH * H_DIM * 4);
    float* gates = (float*)(ws + off); off = align256(off + (size_t)BATCH * G_DIM * 4);
    float* preds = (float*)(ws + off); off = align256(off + (size_t)BATCH * 4);
    float* hp    = (float*)(ws + off); off = align256(off + (size_t)BATCH * 4);
    float* flog  = (float*)(ws + off); off = align256(off + (size_t)BATCH * 4);

    // x in bf16: convert the whole sequence up front if workspace allows,
    // else convert one timestep at a time into a small reusable buffer.
    const size_t xfull_bytes = (size_t)T_STEPS * BATCH * D_IN * 2;
    const bool full_x = (off + xfull_bytes) <= ws_size;
    u16* xb = (u16*)(ws + off);   // full sequence buffer, or one-step buffer

    const long nWih = (long)G_DIM * D_IN;
    const long nWhh = (long)G_DIM * H_DIM;
    const long nXstep = (long)BATCH * D_IN;
    const long nXfull = (long)T_STEPS * nXstep;

    cvt_f32_bf16_kernel<<<(unsigned)((nWih + 255) / 256), 256, 0, stream>>>(W_ih, Wih_bf, nWih);
    cvt_f32_bf16_kernel<<<(unsigned)((nWhh + 255) / 256), 256, 0, stream>>>(W_hh, Whh_bf, nWhh);
    if (full_x) {
        cvt_f32_bf16_kernel<<<(unsigned)((nXfull + 255) / 256), 256, 0, stream>>>(x, xb, nXfull);
    }
    bias_sum_kernel<<<G_DIM / 256, 256, 0, stream>>>(b_ih, b_hh, bias);
    init_state_kernel<<<(BATCH * H_DIM) / 256, 256, 0, stream>>>(hbuf, cbuf, hb, preds, hp);

    for (int t = 0; t < T_STEPS; ++t) {
        const u16* xt;
        if (full_x) {
            xt = xb + (size_t)t * BATCH * D_IN;
        } else {
            cvt_f32_bf16_kernel<<<(unsigned)((nXstep + 255) / 256), 256, 0, stream>>>(
                x + (size_t)t * BATCH * D_IN, xb, nXstep);
            xt = xb;
        }
        // 1024 waves (16 M-tiles x 64 N-groups), 8 waves/block -> 128 blocks
        gates_gemm_kernel<<<128, 256, 0, stream>>>(xt, hb, Wih_bf, Whh_bf, bias, gates);
        lstm_cell_kernel<<<(BATCH * H_DIM) / 256, 256, 0, stream>>>(gates, cbuf, hbuf, hb);
        controller_kernel<<<BATCH / 8, 256, 0, stream>>>(
            hbuf, W_out, b_out, W_ctrl, b_ctrl, u, eps, preds, hp, flog, t);
    }

    finalize_kernel<<<1, BATCH, 0, stream>>>(preds, hp, flog, (float*)d_out);
}
